// AllocationGNN_81441169866931
// MI455X (gfx1250) — compile-verified, matrix operand order checked by reading, not run on hardware
//
#include <hip/hip_runtime.h>
#include <hip/hip_bf16.h>

typedef __attribute__((ext_vector_type(16))) __bf16 v16bf;
typedef __attribute__((ext_vector_type(8)))  float  v8f;

#define NEG_SLOPE 0.2f
#define BM 128        // rows per block  (8 waves x 16)
#define BN 64         // cols per block  (4 WMMA tiles wide)
#define KSTEP 32      // K per WMMA

// ---------- helpers ----------
__device__ __forceinline__ unsigned short f32_to_bf16_rne(float f) {
  unsigned u = __float_as_uint(f);
  unsigned r = u + 0x7FFFu + ((u >> 16) & 1u);
  return (unsigned short)(r >> 16);
}
__device__ __forceinline__ float bf16_to_f32(unsigned short v) {
  return __uint_as_float((unsigned)v << 16);
}
__device__ __forceinline__ unsigned fkey(float f) {       // monotone float->uint
  unsigned u = __float_as_uint(f);
  return (u & 0x80000000u) ? ~u : (u | 0x80000000u);
}
__device__ __forceinline__ float finv(unsigned k) {
  unsigned u = (k & 0x80000000u) ? (k & 0x7FFFFFFFu) : ~k;
  return __uint_as_float(u);
}
__device__ __forceinline__ bool finitef(float f) {
  return (__float_as_uint(f) & 0x7F800000u) != 0x7F800000u;
}
__device__ __forceinline__ void edge_ends(const int* __restrict__ ei, int E, int e,
                                          int& s, int& d) {
  if (e < E) { s = ei[e]; d = ei[E + e]; } else { s = e - E; d = e - E; }  // self loops
}

// CDNA5 async copy: global -> LDS, tracked by ASYNCcnt
__device__ __forceinline__ void async_load_b128(unsigned lds_addr, const void* gaddr) {
  asm volatile("global_load_async_to_lds_b128 %0, %1, off"
               :: "v"(lds_addr), "v"(gaddr) : "memory");
}
__device__ __forceinline__ void wait_async0() {
  asm volatile("s_wait_asynccnt 0x0" ::: "memory");
}
__device__ __forceinline__ void wait_async1() {
  asm volatile("s_wait_asynccnt 0x1" ::: "memory");
}

// ---------- utility kernels ----------
__global__ void k_f32_to_bf16(const float* __restrict__ src,
                              unsigned short* __restrict__ dst, int n) {
  int i = blockIdx.x * blockDim.x + threadIdx.x;
  if (i < n) dst[i] = f32_to_bf16_rne(src[i]);
}
__global__ void k_fill_u32(unsigned* __restrict__ p, unsigned v, int n) {
  int i = blockIdx.x * blockDim.x + threadIdx.x;
  if (i < n) p[i] = v;
}

// ---------- WMMA GEMM with async-LDS staged B tile ----------
// C[M,Nout] = A[M,K] x B[Nout,K]^T ; bf16 in, f32 out + bf16 mirror out.
// Block = 8 waves -> 128x64 output tile; all waves share one 64xKSTEP B slab
// staged through LDS by global_load_async_to_lds_b128, double buffered.
__global__ void __launch_bounds__(256)
k_gemm_bf16_nt(const unsigned short* __restrict__ A,
               const unsigned short* __restrict__ B,
               float* __restrict__ Cf, unsigned short* __restrict__ Cb,
               int M, int Nout, int K) {
  __shared__ __align__(16) unsigned short Bs[2][BN * KSTEP];  // 2 x 4KB

  const int lane = threadIdx.x & 31;
  const int wave = threadIdx.x >> 5;
  const int nBlocksN = Nout / BN;
  const int bm = blockIdx.x / nBlocksN;
  const int bn = blockIdx.x % nBlocksN;
  const int m0 = bm * BM + wave * 16;
  const int n0 = bn * BN;
  const int r  = lane & 15;
  const int kh = (lane >> 4) << 4;                       // K-half: 0 or 16
  const unsigned short* pa = A + (size_t)(m0 + r) * K + kh;

  // stage loader: 256 threads x 16B = 4KB = 64 rows x 32 bf16
  const int ldRow = threadIdx.x >> 2;                    // 0..63
  const int ldCol = (threadIdx.x & 3) << 3;              // 0,8,16,24
  const unsigned short* gB = B + (size_t)(n0 + ldRow) * K + ldCol;
  const unsigned lds0 = (unsigned)(size_t)(&Bs[0][ldRow * KSTEP + ldCol]);
  const unsigned lds1 = (unsigned)(size_t)(&Bs[1][ldRow * KSTEP + ldCol]);

  async_load_b128(lds0, gB);                             // prologue: stage 0

  v8f acc0 = {}, acc1 = {}, acc2 = {}, acc3 = {};
  const int nsteps = K / KSTEP;
  for (int s = 0; s < nsteps; ++s) {
    if (s + 1 < nsteps) {                                // overlap: issue next stage
      async_load_b128((s & 1) ? lds0 : lds1, gB + (s + 1) * KSTEP);
      wait_async1();                                     // oldest (stage s) complete
    } else {
      wait_async0();
    }
    __syncthreads();                                     // cross-wave LDS visibility

    const unsigned short* bs = &Bs[s & 1][0];
    v16bf a  = *(const v16bf*)(const void*)(pa + s * KSTEP);
    v16bf b0 = *(const v16bf*)(const void*)(bs + ( 0 + r) * KSTEP + kh);
    v16bf b1 = *(const v16bf*)(const void*)(bs + (16 + r) * KSTEP + kh);
    v16bf b2 = *(const v16bf*)(const void*)(bs + (32 + r) * KSTEP + kh);
    v16bf b3 = *(const v16bf*)(const void*)(bs + (48 + r) * KSTEP + kh);
    __builtin_prefetch(pa + s * KSTEP + 64, 0, 1);       // global_prefetch_b8
    acc0 = __builtin_amdgcn_wmma_f32_16x16x32_bf16(false, a, false, b0, (short)0, acc0, false, false);
    acc1 = __builtin_amdgcn_wmma_f32_16x16x32_bf16(false, a, false, b1, (short)0, acc1, false, false);
    acc2 = __builtin_amdgcn_wmma_f32_16x16x32_bf16(false, a, false, b2, (short)0, acc2, false, false);
    acc3 = __builtin_amdgcn_wmma_f32_16x16x32_bf16(false, a, false, b3, (short)0, acc3, false, false);
    __syncthreads();                                     // reads done before re-fill
  }

  // C/D layout: lanes 0-15 -> rows m0..m0+7, lanes 16-31 -> rows m0+8..m0+15
  const int mb = m0 + ((lane >> 4) << 3);
  const int nc = n0 + (lane & 15);
#pragma unroll
  for (int i = 0; i < 8; ++i) {
    int m = mb + i;
    if (m < M) {
      float* c = Cf + (size_t)m * Nout + nc;
      c[0] = acc0[i]; c[16] = acc1[i]; c[32] = acc2[i]; c[48] = acc3[i];
      unsigned short* cb = Cb + (size_t)m * Nout + nc;   // bf16 mirror for gathers
      cb[0]  = f32_to_bf16_rne(acc0[i]);
      cb[16] = f32_to_bf16_rne(acc1[i]);
      cb[32] = f32_to_bf16_rne(acc2[i]);
      cb[48] = f32_to_bf16_rne(acc3[i]);
    }
  }
}

// ---------- per-node attention scalars: s_n, d_n ----------
__global__ void k_attn_sd(const float* __restrict__ h, const float* __restrict__ a_src,
                          const float* __restrict__ a_dst, float* __restrict__ s,
                          float* __restrict__ d, int N, int H, int C) {
  int i = blockIdx.x * blockDim.x + threadIdx.x;
  if (i >= N * H) return;
  int n = i / H, hh = i % H;
  const float* hp = h + (size_t)n * H * C + (size_t)hh * C;
  const float* as = a_src + hh * C;
  const float* ad = a_dst + hh * C;
  float ss = 0.f, dd = 0.f;
  for (int c = 0; c < C; ++c) { float v = hp[c]; ss += v * as[c]; dd += v * ad[c]; }
  s[i] = ss; d[i] = dd;
}

// ---------- edge passes ----------
__global__ void k_edge_max(const float* __restrict__ sn, const float* __restrict__ dn,
                           const int* __restrict__ ei, int E, int Et, int H,
                           float* __restrict__ e_buf, unsigned* __restrict__ mkey) {
  int i = blockIdx.x * blockDim.x + threadIdx.x;
  if (i >= Et * H) return;
  int e = i / H, hh = i % H;
  int s, d; edge_ends(ei, E, e, s, d);
  float v = sn[s * H + hh] + dn[d * H + hh];
  v = v > 0.f ? v : NEG_SLOPE * v;                   // leaky_relu
  e_buf[i] = v;
  atomicMax(&mkey[d * H + hh], fkey(v));             // segment max
}

__global__ void k_edge_exp(float* __restrict__ e_buf, const unsigned* __restrict__ mkey,
                           const int* __restrict__ ei, int E, int Et, int H,
                           float* __restrict__ denom) {
  int i = blockIdx.x * blockDim.x + threadIdx.x;
  if (i >= Et * H) return;
  int e = i / H, hh = i % H;
  int s, d; edge_ends(ei, E, e, s, d);
  float m = finv(mkey[d * H + hh]);
  if (!finitef(m)) m = 0.f;
  float ex = __expf(e_buf[i] - m);
  e_buf[i] = ex;
  atomicAdd(&denom[d * H + hh], ex);                 // segment sum
}

// block per edge; gathers bf16 payload (half the read traffic), accumulates f32
__global__ void k_edge_msg(const unsigned short* __restrict__ hb,
                           const float* __restrict__ ex,
                           const float* __restrict__ denom, const int* __restrict__ ei,
                           int E, int Et, int H, int C, float* __restrict__ agg) {
  int e = blockIdx.x;
  if (e >= Et) return;
  int s, d; edge_ends(ei, E, e, s, d);
  int HC = H * C;
  for (int c = threadIdx.x; c < HC; c += blockDim.x) {
    int hh = c / C;
    float a = ex[e * H + hh] / (denom[d * H + hh] + 1e-16f);
    float hv = bf16_to_f32(hb[(size_t)s * HC + c]);
    atomicAdd(&agg[(size_t)d * HC + c], hv * a);
  }
}

// ---------- epilogues ----------
__global__ void k_bias_elu_bf16(const float* __restrict__ agg, const float* __restrict__ b,
                                unsigned short* __restrict__ out, int total, int HC) {
  int i = blockIdx.x * blockDim.x + threadIdx.x;
  if (i >= total) return;
  float v = agg[i] + b[i % HC];
  v = v > 0.f ? v : (__expf(v) - 1.f);               // elu
  out[i] = f32_to_bf16_rne(v);
}
__global__ void k_bias_f32(const float* __restrict__ agg, const float* __restrict__ b,
                           float* __restrict__ out, int total, int HC) {
  int i = blockIdx.x * blockDim.x + threadIdx.x;
  if (i >= total) return;
  out[i] = agg[i] + b[i % HC];
}

// ---------- allocation head ----------
__global__ void k_mlp1(const float* __restrict__ emb, const float* __restrict__ W,
                       const float* __restrict__ b, float* __restrict__ z, int N) {
  int i = blockIdx.x * blockDim.x + threadIdx.x;     // z[n,j], j<64
  if (i >= N * 64) return;
  int n = i / 64, j = i % 64;
  const float* ep = emb + (size_t)n * 128;
  const float* wp = W + j * 128;
  float acc = b[j];
  for (int c = 0; c < 128; ++c) acc += ep[c] * wp[c];
  z[i] = acc > 0.f ? acc : 0.f;
}
__global__ void k_mlp2(const float* __restrict__ z, const float* __restrict__ W,
                       const float* __restrict__ b, float* __restrict__ out, int N) {
  int n = blockIdx.x * blockDim.x + threadIdx.x;
  if (n >= N) return;
  const float* zp = z + (size_t)n * 64;
  float acc = b[0];
  for (int c = 0; c < 64; ++c) acc += zp[c] * W[c];
  out[n] = 1.f / (1.f + __expf(-acc));
}

// ---------- host orchestration ----------
extern "C" void kernel_launch(void* const* d_in, const int* in_sizes, int n_in,
                              void* d_out, int out_size, void* d_ws, size_t ws_size,
                              hipStream_t stream) {
  (void)n_in; (void)out_size; (void)ws_size;
  const float* x   = (const float*)d_in[0];
  const int*   ei  = (const int*)d_in[1];
  const float* W1  = (const float*)d_in[2];
  const float* as1 = (const float*)d_in[3];
  const float* ad1 = (const float*)d_in[4];
  const float* b1  = (const float*)d_in[5];
  const float* W2  = (const float*)d_in[6];
  const float* as2 = (const float*)d_in[7];
  const float* ad2 = (const float*)d_in[8];
  const float* b2  = (const float*)d_in[9];
  const float* W3  = (const float*)d_in[10];
  const float* as3 = (const float*)d_in[11];
  const float* ad3 = (const float*)d_in[12];
  const float* b3  = (const float*)d_in[13];
  const float* Wh1 = (const float*)d_in[14];
  const float* bh1 = (const float*)d_in[15];
  const float* Wh2 = (const float*)d_in[16];
  const float* bh2 = (const float*)d_in[17];

  const int F = 256, C = 128, HCmax = 512;
  const int N  = in_sizes[0] / F;
  const int E  = in_sizes[1] / 2;
  const int Et = E + N;
  const int Mpad = (N + BM - 1) & ~(BM - 1);   // A padded to block-tile height

  // workspace carve-up
  char* ws = (char*)d_ws;
  size_t off = 0;
  auto carve = [&](size_t bytes) -> void* {
    void* p = ws + off; off += (bytes + 255) & ~(size_t)255; return p;
  };
  unsigned short* Abuf = (unsigned short*)carve((size_t)Mpad * HCmax * 2); // bf16 activations
  unsigned short* Wbuf = (unsigned short*)carve((size_t)HCmax * HCmax * 2);// bf16 weights
  float*          Hbuf = (float*)carve((size_t)N * HCmax * 4);             // GEMM out h (f32)
  unsigned short* Hb   = (unsigned short*)carve((size_t)N * HCmax * 2);    // bf16 mirror
  float*    Agg  = (float*)carve((size_t)N * HCmax * 4);                   // scatter target
  float*    Sbuf = (float*)carve((size_t)N * 4 * 4);
  float*    Dbuf = (float*)carve((size_t)N * 4 * 4);
  unsigned* MK   = (unsigned*)carve((size_t)N * 4 * 4);
  float*    Den  = (float*)carve((size_t)N * 4 * 4);
  float*    Ebuf = (float*)carve((size_t)Et * 4 * 4);                      // edge logits / ex
  float*    Zbuf = (float*)carve((size_t)N * 64 * 4);                      // MLP hidden

  float* emb    = (float*)d_out;            // [N,128]
  float* scores = emb + (size_t)N * C;      // [N]

  const int TB = 256;
  auto cdiv = [](long long a, long long b) { return (unsigned)((a + b - 1) / b); };

  // x -> bf16
  k_f32_to_bf16<<<cdiv((long long)N * F, TB), TB, 0, stream>>>(x, Abuf, N * F);

  struct Layer { const float *W, *as, *ad, *b; int Kin, Hh, Cc; };
  Layer L[3] = { {W1, as1, ad1, b1, F,   4, C},
                 {W2, as2, ad2, b2, 512, 4, C},
                 {W3, as3, ad3, b3, 512, 1, C} };

  for (int li = 0; li < 3; ++li) {
    const int K = L[li].Kin, Hh = L[li].Hh, Cc = L[li].Cc, HC = Hh * Cc;
    // weights -> bf16
    k_f32_to_bf16<<<cdiv((long long)HC * K, TB), TB, 0, stream>>>(L[li].W, Wbuf, HC * K);
    // dense transform on the matrix pipes (async-LDS staged)
    {
      unsigned blocks = (unsigned)((Mpad / BM) * (HC / BN));
      k_gemm_bf16_nt<<<blocks, TB, 0, stream>>>(Abuf, Wbuf, Hbuf, Hb, N, HC, K);
    }
    // attention scalars
    k_attn_sd<<<cdiv((long long)N * Hh, TB), TB, 0, stream>>>(Hbuf, L[li].as, L[li].ad,
                                                              Sbuf, Dbuf, N, Hh, Cc);
    // reduction buffers
    k_fill_u32<<<cdiv((long long)N * Hh, TB), TB, 0, stream>>>(MK, 0u, N * Hh);
    k_fill_u32<<<cdiv((long long)N * Hh, TB), TB, 0, stream>>>((unsigned*)Den, 0u, N * Hh);
    k_fill_u32<<<cdiv((long long)N * HC, TB), TB, 0, stream>>>((unsigned*)Agg, 0u, N * HC);
    // segment softmax + message aggregation (bandwidth-bound phase)
    k_edge_max<<<cdiv((long long)Et * Hh, TB), TB, 0, stream>>>(Sbuf, Dbuf, ei, E, Et, Hh, Ebuf, MK);
    k_edge_exp<<<cdiv((long long)Et * Hh, TB), TB, 0, stream>>>(Ebuf, MK, ei, E, Et, Hh, Den);
    k_edge_msg<<<dim3(Et), dim3(128), 0, stream>>>(Hb, Ebuf, Den, ei, E, Et, Hh, Cc, Agg);
    // epilogue
    if (li < 2)
      k_bias_elu_bf16<<<cdiv((long long)N * HC, TB), TB, 0, stream>>>(Agg, L[li].b, Abuf, N * HC, HC);
    else
      k_bias_f32<<<cdiv((long long)N * HC, TB), TB, 0, stream>>>(Agg, L[li].b, emb, N * HC, HC);
  }

  // allocation head
  k_mlp1<<<cdiv((long long)N * 64, TB), TB, 0, stream>>>(emb, Wh1, bh1, Zbuf, N);
  k_mlp2<<<cdiv((long long)N, TB), TB, 0, stream>>>(Zbuf, Wh2, bh2, scores, N);
}
